// GNNNet_23244363006304
// MI455X (gfx1250) — matrix operand order repaired and checked by reading, not compiled, not run on hardware
//
#include <hip/hip_runtime.h>
#include <hip/hip_fp16.h>

#define NSLOPE 0.2f

typedef __attribute__((ext_vector_type(16))) _Float16 v16h;
typedef __attribute__((ext_vector_type(8)))  float    v8f;
typedef __attribute__((ext_vector_type(4)))  int      v4i;

#if defined(__gfx1250__) && __has_builtin(__builtin_amdgcn_global_load_async_to_lds_b128)
#define HAS_ASYNC_LDS 1
#else
#define HAS_ASYNC_LDS 0
#endif

#if HAS_ASYNC_LDS
typedef __attribute__((address_space(1))) v4i* gvec_ptr;
typedef __attribute__((address_space(3))) v4i* lvec_ptr;
#endif

// ------------------------------------------------------------------
// helpers: monotone float<->uint encoding for atomic max on floats
// ------------------------------------------------------------------
__device__ __forceinline__ unsigned fenc(float f) {
    unsigned b = __float_as_uint(f);
    return (b & 0x80000000u) ? ~b : (b | 0x80000000u);
}
__device__ __forceinline__ float fdec(unsigned e) {
    unsigned b = (e & 0x80000000u) ? (e & 0x7FFFFFFFu) : ~e;
    return __uint_as_float(b);
}

// ------------------------------------------------------------------
// Pack f32 row-major W[K x N] into CDNA5 WMMA B-operand tiles (f16).
// Layout per (kb, nt) tile: lane l in [0,32): n = nt*16 + (l&15),
// K-half = l>>4; halves h in [0,16): K = kb*32 + (l>>4)*16 + h.
// Flat index: (((kb*ntiles + nt)*32) + l)*16 + h
// ------------------------------------------------------------------
__global__ void pack_b_kernel(const float* __restrict__ W,
                              _Float16* __restrict__ out, int K, int N) {
    int total = K * N;
    int ntiles = N >> 4;
    for (int t = blockIdx.x * blockDim.x + threadIdx.x; t < total;
         t += gridDim.x * blockDim.x) {
        int h    = t & 15;
        int l    = (t >> 4) & 31;
        int tile = t >> 9;
        int nt   = tile % ntiles;
        int kb   = tile / ntiles;
        int k = kb * 32 + (l >> 4) * 16 + h;
        int n = nt * 16 + (l & 15);
        out[t] = (_Float16)W[(size_t)k * N + n];
    }
}

// ------------------------------------------------------------------
// WMMA GEMM: C[M x N] = act(A[M x K] * W + bias)
// one wave per 16x16 tile; blockDim.x = 32*(N/16); gridDim.x = M/16.
// The 16 x K f32 A-tile is staged once per block into LDS via the
// gfx1250 async global->LDS path (ASYNCcnt), then all waves build
// f16 A fragments from LDS. Row stride padded to K+4 floats to
// spread the 16 half-wave lanes across the 64 LDS banks.
// act: 0 = none, 1 = relu, 2 = tanh
// ------------------------------------------------------------------
__global__ void gemm_wmma_kernel(const float* __restrict__ A,
                                 const _Float16* __restrict__ Bp,
                                 const float* __restrict__ bias,
                                 float* __restrict__ C,
                                 int M, int N, int K, int act) {
    __shared__ float sA[16 * (128 + 4)];   // max K = 128, padded stride

    const int tid      = threadIdx.x;
    const int nthreads = blockDim.x;
    const int mt       = blockIdx.x;
    const int stride   = K + 4;            // padded LDS row stride (floats)
    const int vecPerRow = K >> 2;          // float4 chunks per row

    // ---- stage A tile (16 rows x K) into LDS ----
    for (int i = tid; i < 16 * vecPerRow; i += nthreads) {
        int r = i / vecPerRow;
        int c = i - r * vecPerRow;
        int grow = mt * 16 + r;
        if (grow >= M) grow = M - 1;       // clamp (keeps addresses valid)
        const float* src = A + (size_t)grow * K + c * 4;
        float*       dst = sA + r * stride + c * 4;
#if HAS_ASYNC_LDS
        __builtin_amdgcn_global_load_async_to_lds_b128(
            (gvec_ptr)src, (lvec_ptr)dst, 0, 0);
#else
        *(float4*)dst = *(const float4*)src;
#endif
    }
#if HAS_ASYNC_LDS
    __builtin_amdgcn_s_wait_asynccnt(0);
#endif
    __syncthreads();

    const int lane  = tid & 31;
    const int wave  = tid >> 5;
    const int nt    = wave;
    const int khalf = lane >> 4;
    const int ntiles = N >> 4;

    v8f acc = {0.f, 0.f, 0.f, 0.f, 0.f, 0.f, 0.f, 0.f};
    const int nkb = K >> 5;
    for (int kb = 0; kb < nkb; ++kb) {
        // A lane layout used here (contiguous k, as packed B): lanes 0-15 hold K 0-15,
        // lanes 16-31 hold K 16-31 of row M = lane%16.
        const float* arow = sA + (lane & 15) * stride + kb * 32 + khalf * 16;
        v16h a;
#pragma unroll
        for (int j = 0; j < 8; ++j) a[j]     = (_Float16)arow[j];
#pragma unroll
        for (int j = 0; j < 8; ++j) a[8 + j] = (_Float16)arow[8 + j];

        v16h b = *(const v16h*)(Bp + (((size_t)kb * ntiles + nt) * 32 + lane) * 16);

        acc = __builtin_amdgcn_wmma_f32_16x16x32_f16(
            false, a, false, b, (short)0, acc, false, false);
    }

    // C/D layout: lane holds column N = lane%16; VGPR r -> M = r + 8*(lane/16)
    int col   = nt * 16 + (lane & 15);
    float bv  = bias ? bias[col] : 0.0f;
    int mbase = mt * 16 + khalf * 8;
#pragma unroll
    for (int r = 0; r < 8; ++r) {
        int row = mbase + r;
        if (row < M) {
            float v = acc[r] + bv;
            if (act == 1)      v = fmaxf(v, 0.0f);
            else if (act == 2) v = tanhf(v);
            C[(size_t)row * N + col] = v;
        }
    }
}

// ------------------------------------------------------------------
// Edge score: s[e] = sum_k att[k] * leakyrelu(XL[src,k] + XR[dst,k])
// one wave per edge; 4 dims per lane; wave reduction.
// ------------------------------------------------------------------
__global__ void edge_score_kernel(const float* __restrict__ XL,
                                  const float* __restrict__ XR,
                                  const float* __restrict__ att,
                                  const int* __restrict__ ei,
                                  int E, int Nn, float* __restrict__ s) {
    int gw   = (int)((blockIdx.x * blockDim.x + threadIdx.x) >> 5);
    int lane = threadIdx.x & 31;
    int Etot = E + Nn;
    if (gw >= Etot) return;
    int src, dst;
    if (gw < E) { src = ei[gw]; dst = ei[E + gw]; }
    else        { src = dst = gw - E; }
    const float* xl = XL + (size_t)src * 128;
    const float* xr = XR + (size_t)dst * 128;
    float acc = 0.f;
#pragma unroll
    for (int j = 0; j < 4; ++j) {
        int k   = lane + 32 * j;
        float v = xl[k] + xr[k];
        v = v > 0.f ? v : NSLOPE * v;
        acc += att[k] * v;
    }
#pragma unroll
    for (int off = 16; off > 0; off >>= 1) acc += __shfl_xor(acc, off, 32);
    if (lane == 0) s[gw] = acc;
}

// segment max over dst (uint-encoded float atomicMax; init 0 == -huge)
__global__ void seg_max_kernel(const float* __restrict__ s,
                               const int* __restrict__ ei,
                               int E, int Nn, unsigned* __restrict__ menc) {
    int e = blockIdx.x * blockDim.x + threadIdx.x;
    int Etot = E + Nn;
    if (e >= Etot) return;
    int dst = (e < E) ? ei[E + e] : e - E;
    atomicMax(&menc[dst], fenc(s[e]));
}

// ex = exp(s - m[dst]); denom[dst] += ex
__global__ void seg_exp_kernel(const float* __restrict__ s,
                               const int* __restrict__ ei,
                               int E, int Nn,
                               const unsigned* __restrict__ menc,
                               float* __restrict__ ex,
                               float* __restrict__ denom) {
    int e = blockIdx.x * blockDim.x + threadIdx.x;
    int Etot = E + Nn;
    if (e >= Etot) return;
    int dst  = (e < E) ? ei[E + e] : e - E;
    float v  = expf(s[e] - fdec(menc[dst]));
    ex[e] = v;
    atomicAdd(&denom[dst], v);
}

// out[dst] += alpha * XL[src]; one wave per edge, 4 dims per lane
__global__ void scatter_kernel(const float* __restrict__ XL,
                               const int* __restrict__ ei,
                               int E, int Nn,
                               const float* __restrict__ ex,
                               const float* __restrict__ denom,
                               float* __restrict__ out) {
    int gw   = (int)((blockIdx.x * blockDim.x + threadIdx.x) >> 5);
    int lane = threadIdx.x & 31;
    int Etot = E + Nn;
    if (gw >= Etot) return;
    int src, dst;
    if (gw < E) { src = ei[gw]; dst = ei[E + gw]; }
    else        { src = dst = gw - E; }
    float alpha = ex[gw] / denom[dst];
    const float* xl = XL + (size_t)src * 128;
    float*       o  = out + (size_t)dst * 128;
#pragma unroll
    for (int j = 0; j < 4; ++j) {
        int k = lane + 32 * j;
        atomicAdd(&o[k], alpha * xl[k]);
    }
}

// h = relu(h + b[k])
__global__ void bias_relu_kernel(float* __restrict__ h,
                                 const float* __restrict__ b, size_t total) {
    for (size_t t = (size_t)blockIdx.x * blockDim.x + threadIdx.x; t < total;
         t += (size_t)gridDim.x * blockDim.x) {
        h[t] = fmaxf(h[t] + b[t & 127], 0.0f);
    }
}

// value head: partial += sum_i dot(h4[i,0:64], WV)
__global__ void value_kernel(const float* __restrict__ h4,
                             const float* __restrict__ WV,
                             int Nn, float* __restrict__ partial) {
    float local = 0.f;
    for (int i = blockIdx.x * blockDim.x + threadIdx.x; i < Nn;
         i += gridDim.x * blockDim.x) {
        const float* r = h4 + (size_t)i * 64;
        float d = 0.f;
#pragma unroll
        for (int k = 0; k < 64; ++k) d += r[k] * WV[k];
        local += d;
    }
#pragma unroll
    for (int off = 16; off > 0; off >>= 1) local += __shfl_xor(local, off, 32);
    if ((threadIdx.x & 31) == 0) atomicAdd(partial, local);
}

__global__ void value_fin_kernel(const float* __restrict__ partial,
                                 const float* __restrict__ bV,
                                 int Nn, float* __restrict__ out) {
    if (blockIdx.x == 0 && threadIdx.x == 0)
        out[0] = partial[0] / (float)Nn + bV[0];
}

// ------------------------------------------------------------------
extern "C" void kernel_launch(void* const* d_in, const int* in_sizes, int n_in,
                              void* d_out, int out_size, void* d_ws, size_t ws_size,
                              hipStream_t stream) {
    const float* x     = (const float*)d_in[0];
    const int*   ei    = (const int*)d_in[1];
    const float* Wl1   = (const float*)d_in[3];
    const float* bl1   = (const float*)d_in[4];
    const float* Wr1   = (const float*)d_in[5];
    const float* br1   = (const float*)d_in[6];
    const float* att1  = (const float*)d_in[7];
    const float* b1    = (const float*)d_in[8];
    const float* Wl2   = (const float*)d_in[9];
    const float* bl2   = (const float*)d_in[10];
    const float* Wr2   = (const float*)d_in[11];
    const float* br2   = (const float*)d_in[12];
    const float* att2  = (const float*)d_in[13];
    const float* b2    = (const float*)d_in[14];
    const float* Wlin1 = (const float*)d_in[15];
    const float* blin1 = (const float*)d_in[16];
    const float* Wlin2 = (const float*)d_in[17];
    const float* blin2 = (const float*)d_in[18];
    const float* Wlin3 = (const float*)d_in[19];
    const float* blin3 = (const float*)d_in[20];
    const float* WV    = (const float*)d_in[21];
    const float* bV    = (const float*)d_in[22];

    const int Nn   = in_sizes[0] / 128;   // 100000
    const int E    = in_sizes[1] / 2;     // 600000
    const int Etot = E + Nn;              // 700000

    // ---- workspace layout (256B aligned chunks) ----
    char* ws = (char*)d_ws;
    size_t off = 0;
    auto alloc = [&](size_t bytes) -> char* {
        char* p = ws + off;
        off += (bytes + 255) & ~(size_t)255;
        return p;
    };
    const size_t NB = (size_t)Nn * 128 * sizeof(float);
    float*    bufA  = (float*)alloc(NB);            // XL / lin outputs
    float*    bufB  = (float*)alloc(NB);            // XR / h2
    float*    bufC  = (float*)alloc(NB);            // h1 / h3
    float*    S     = (float*)alloc((size_t)Etot * 4);
    float*    EX    = (float*)alloc((size_t)Etot * 4);
    unsigned* Menc  = (unsigned*)alloc((size_t)Nn * 4);
    float*    Den   = (float*)alloc((size_t)Nn * 4);
    float*    Val   = (float*)alloc(4);
    _Float16* pWl1  = (_Float16*)alloc(128 * 128 * 2);
    _Float16* pWr1  = (_Float16*)alloc(128 * 128 * 2);
    _Float16* pWl2  = (_Float16*)alloc(128 * 128 * 2);
    _Float16* pWr2  = (_Float16*)alloc(128 * 128 * 2);
    _Float16* pWlin1 = (_Float16*)alloc(128 * 128 * 2);
    _Float16* pWlin2 = (_Float16*)alloc(128 * 64 * 2);
    _Float16* pWlin3 = (_Float16*)alloc(64 * 16 * 2);

    // ---- pack weights into WMMA B layout ----
    pack_b_kernel<<<64, 256, 0, stream>>>(Wl1,   pWl1,   128, 128);
    pack_b_kernel<<<64, 256, 0, stream>>>(Wr1,   pWr1,   128, 128);
    pack_b_kernel<<<64, 256, 0, stream>>>(Wl2,   pWl2,   128, 128);
    pack_b_kernel<<<64, 256, 0, stream>>>(Wr2,   pWr2,   128, 128);
    pack_b_kernel<<<64, 256, 0, stream>>>(Wlin1, pWlin1, 128, 128);
    pack_b_kernel<<<64, 256, 0, stream>>>(Wlin2, pWlin2, 128, 64);
    pack_b_kernel<<<8,  256, 0, stream>>>(Wlin3, pWlin3, 64, 16);

    auto gemm = [&](const float* A, const _Float16* Bp, const float* bias,
                    float* C, int M, int N, int K, int act) {
        dim3 blk(32 * (N / 16));
        dim3 grd((M + 15) / 16);
        gemm_wmma_kernel<<<grd, blk, 0, stream>>>(A, Bp, bias, C, M, N, K, act);
    };

    const int edgeThreadBlocks = (Etot + 255) / 256;
    const int edgeWaveBlocks   = (Etot + 7) / 8;     // 8 waves per 256-thread block

    auto gat_layer = [&](const float* X, const _Float16* pWl, const float* bl,
                         const _Float16* pWr, const float* br,
                         const float* att, const float* bout, float* H) {
        gemm(X, pWl, bl, bufA, Nn, 128, 128, 0);   // XL
        gemm(X, pWr, br, bufB, Nn, 128, 128, 0);   // XR
        edge_score_kernel<<<edgeWaveBlocks, 256, 0, stream>>>(bufA, bufB, att,
                                                              ei, E, Nn, S);
        (void)hipMemsetAsync(Menc, 0, (size_t)Nn * 4, stream);
        (void)hipMemsetAsync(Den,  0, (size_t)Nn * 4, stream);
        seg_max_kernel<<<edgeThreadBlocks, 256, 0, stream>>>(S, ei, E, Nn, Menc);
        seg_exp_kernel<<<edgeThreadBlocks, 256, 0, stream>>>(S, ei, E, Nn, Menc,
                                                             EX, Den);
        (void)hipMemsetAsync(H, 0, NB, stream);
        scatter_kernel<<<edgeWaveBlocks, 256, 0, stream>>>(bufA, ei, E, Nn,
                                                           EX, Den, H);
        bias_relu_kernel<<<4096, 256, 0, stream>>>(H, bout, (size_t)Nn * 128);
    };

    // layer 1: x -> bufC ; layer 2: bufC -> bufB
    gat_layer(x,    pWl1, bl1, pWr1, br1, att1, b1, bufC);
    gat_layer(bufC, pWl2, bl2, pWr2, br2, att2, b2, bufB);

    // lin1: bufB -> bufC (no activation)
    gemm(bufB, pWlin1, blin1, bufC, Nn, 128, 128, 0);
    // lin2: bufC -> bufA [N,64]
    gemm(bufC, pWlin2, blin2, bufA, Nn, 64, 128, 0);
    // out = tanh(h4 @ Wlin3 + blin3) -> d_out[0 .. Nn*16)
    gemm(bufA, pWlin3, blin3, (float*)d_out, Nn, 16, 64, 2);

    // value = mean(h4 @ WV) + bV -> d_out[Nn*16]
    (void)hipMemsetAsync(Val, 0, 4, stream);
    value_kernel<<<256, 256, 0, stream>>>(bufA, WV, Nn, Val);
    value_fin_kernel<<<1, 32, 0, stream>>>(Val, bV, Nn, (float*)d_out + (size_t)Nn * 16);
}
